// SelfAttention_3169685864899
// MI455X (gfx1250) — compile-verified, matrix operand order checked by reading, not run on hardware
//
#include <hip/hip_runtime.h>
#include <hip/hip_bf16.h>

typedef __bf16 bf16;
typedef __attribute__((ext_vector_type(16))) __bf16 v16bf;
typedef __attribute__((ext_vector_type(8)))  __bf16 v8bf;
typedef __attribute__((ext_vector_type(8)))  float  v8f;

#define INPUT_DIM 1024
#define HEAD_DIM  64
#define HEAD_NUM  16
#define INNER     1024
#define SEQ       2048
#define BATCH     4
#define TOKENS    (BATCH * SEQ)   // 8192

static __device__ __forceinline__ v8f wmma_bf16(v16bf a, v16bf b, v8f c) {
  // D = A(16x32 bf16) x B(32x16 bf16) + C(16x16 f32)
  return __builtin_amdgcn_wmma_f32_16x16x32_bf16(
      /*neg_a=*/false, a, /*neg_b=*/false, b,
      /*c_mod=*/(short)0, c, /*reuse_a=*/false, /*reuse_b=*/false);
}

// ---------------------------------------------------------------------------
// fp32 -> bf16 conversion (grid-stride)
// ---------------------------------------------------------------------------
__global__ void cvt_f32_bf16(const float* __restrict__ in, bf16* __restrict__ out, int n) {
  int i = blockIdx.x * blockDim.x + threadIdx.x;
  int stride = gridDim.x * blockDim.x;
  for (; i < n; i += stride) out[i] = (bf16)in[i];
}

// ---------------------------------------------------------------------------
// QKV GEMM: C[8192,3072] = X[8192,1024] x Wqkv[3072,1024]^T
// Q,K columns (n < 2048) stored row-major in QK[8192,2048] (bf16).
// V columns stored TRANSPOSED into VT[b,h,d,t] so the PV matmul's B operand
// is key-contiguous later.
// Block: 256 thr = 8 waves (2 M x 4 N); wave tile 64x32 (4x2 WMMA accum);
// block tile 128x128.  8 WMMAs per 6KB of fragment loads (~21 FLOP/B).
// ---------------------------------------------------------------------------
__global__ void qkv_gemm(const bf16* __restrict__ A,   // [8192,1024]
                         const bf16* __restrict__ W,   // [3072,1024]
                         bf16* __restrict__ QK,        // [8192,2048]
                         bf16* __restrict__ VT) {      // [4,16,64,2048]
  const int K    = INPUT_DIM;
  const int lane = threadIdx.x & 31;
  const int wave = threadIdx.x >> 5;
  const int m0 = blockIdx.x * 128 + (wave >> 2) * 64;   // 2 waves along M
  const int n0 = blockIdx.y * 128 + (wave & 3) * 32;    // 4 waves along N
  const int r    = lane & 15;
  const int hl   = lane >> 4;
  const int koff = hl * 16;

  const bf16* aB = A + (size_t)(m0 + r) * K + koff;
  const bf16* bB = W + (size_t)(n0 + r) * K + koff;

  v8f acc[4][2] = {};
  for (int k = 0; k < K; k += 32) {
    __builtin_prefetch(aB + k + 256, 0, 3);   // WGP-scope, ~8 k-steps ahead
    __builtin_prefetch(bB + k + 256, 0, 3);
    v16bf bF0 = *(const v16bf*)(bB + k);
    v16bf bF1 = *(const v16bf*)(bB + (size_t)16 * K + k);
#pragma unroll
    for (int i = 0; i < 4; ++i) {
      v16bf aF = *(const v16bf*)(aB + (size_t)(16 * i) * K + k);
      acc[i][0] = wmma_bf16(aF, bF0, acc[i][0]);
      acc[i][1] = wmma_bf16(aF, bF1, acc[i][1]);
    }
  }

  const int cn = r;
#pragma unroll
  for (int mt = 0; mt < 4; ++mt) {
#pragma unroll
    for (int nt = 0; nt < 2; ++nt) {
#pragma unroll
      for (int rr = 0; rr < 8; ++rr) {
        int m = m0 + mt * 16 + rr + 8 * hl;
        int n = n0 + nt * 16 + cn;
        bf16 v = (bf16)acc[mt][nt][rr];
        if (n < 2 * INNER) {                     // Q or K: row-major
          QK[(size_t)m * (2 * INNER) + n] = v;
        } else {                                 // V: store transposed
          int c2 = n - 2 * INNER;
          int h = c2 >> 6, d = c2 & 63;
          int bb = m >> 11, t = m & (SEQ - 1);
          VT[(((size_t)bb * HEAD_NUM + h) * HEAD_DIM + d) * SEQ + t] = v;
        }
      }
    }
  }
}

// ---------------------------------------------------------------------------
// Flash-style attention: one wave = one (b,h, 32-query tile).
// Per iteration: 32 keys -> S(32x32) via 8 WMMAs, online softmax,
// P bounced through LDS (C-layout -> A-layout), 8 WMMAs for P x V.
// Every K/V fragment is reused across both query M-tiles (2x intensity).
// ---------------------------------------------------------------------------
__global__ void attn_fa(const bf16* __restrict__ QK,  // [8192,2048] (q | k)
                        const bf16* __restrict__ VT,  // [b,h,64,2048]
                        bf16* __restrict__ AO) {      // [8192,1024]
  __shared__ __attribute__((aligned(16))) bf16 pbuf[8][32][40]; // 20.5KB, stride 80B
  const int lane = threadIdx.x & 31;
  const int wv   = threadIdx.x >> 5;
  const int gw    = blockIdx.x * 8 + wv;
  const int mtile = gw & 63;             // 64 tiles of 32 queries per (b,h)
  const int bh    = gw >> 6;             // 0..63
  const int b = bh >> 4, h = bh & 15;
  const int r = lane & 15, hl = lane >> 4, cn = r;
  const int koff = hl * 16;
  const size_t ldq = 2 * INNER;          // 2048

  const bf16* qrow = QK + (size_t)(b * SEQ + mtile * 32 + r) * ldq + h * HEAD_DIM + koff;
  v16bf qf[2][2];
  qf[0][0] = *(const v16bf*)(qrow);
  qf[0][1] = *(const v16bf*)(qrow + 32);
  qf[1][0] = *(const v16bf*)(qrow + (size_t)16 * ldq);
  qf[1][1] = *(const v16bf*)(qrow + (size_t)16 * ldq + 32);

  const bf16* kbase = QK + (size_t)(b * SEQ) * ldq + INNER + h * HEAD_DIM + koff;
  const bf16* vbase = VT + (size_t)bh * HEAD_DIM * SEQ;

  const float scale = 0.125f;            // 1/sqrt(64)
  v8f o[2][4] = {};
  float mrow[2][8], lrow[2][8];
#pragma unroll
  for (int mt = 0; mt < 2; ++mt)
#pragma unroll
    for (int i = 0; i < 8; ++i) { mrow[mt][i] = -1e30f; lrow[mt][i] = 0.f; }

  for (int kb = 0; kb < SEQ; kb += 32) {
    // ---- scores S = Q K^T for 32 keys (two 16-key N tiles, two M tiles) ----
    v8f s[2][2] = {};
    {
      const bf16* kr0 = kbase + (size_t)(kb + r) * ldq;
      v16bf kA0 = *(const v16bf*)(kr0);
      v16bf kA1 = *(const v16bf*)(kr0 + 32);
      v16bf kB0 = *(const v16bf*)(kr0 + (size_t)16 * ldq);
      v16bf kB1 = *(const v16bf*)(kr0 + (size_t)16 * ldq + 32);
#pragma unroll
      for (int mt = 0; mt < 2; ++mt) {
        s[mt][0] = wmma_bf16(qf[mt][0], kA0, s[mt][0]);
        s[mt][0] = wmma_bf16(qf[mt][1], kA1, s[mt][0]);
        s[mt][1] = wmma_bf16(qf[mt][0], kB0, s[mt][1]);
        s[mt][1] = wmma_bf16(qf[mt][1], kB1, s[mt][1]);
      }
    }
    // ---- online softmax per query row ----
#pragma unroll
    for (int mt = 0; mt < 2; ++mt) {
#pragma unroll
      for (int rr = 0; rr < 8; ++rr) {
        float sa = s[mt][0][rr] * scale;
        float sb = s[mt][1][rr] * scale;
        float mloc = fmaxf(sa, sb);
#pragma unroll
        for (int d = 8; d >= 1; d >>= 1)
          mloc = fmaxf(mloc, __shfl_xor(mloc, d, 32));
        float mnew = fmaxf(mrow[mt][rr], mloc);
        float corr = __expf(mrow[mt][rr] - mnew);
        float p0 = __expf(sa - mnew);
        float p1 = __expf(sb - mnew);
        float ps = p0 + p1;
#pragma unroll
        for (int d = 8; d >= 1; d >>= 1)
          ps += __shfl_xor(ps, d, 32);
        lrow[mt][rr] = lrow[mt][rr] * corr + ps;
        mrow[mt][rr] = mnew;
#pragma unroll
        for (int j = 0; j < 4; ++j) o[mt][j][rr] *= corr;
        // C-layout element (row mt*16+rr+8*hl, keys kb+cn / kb+16+cn)
        pbuf[wv][mt * 16 + rr + 8 * hl][cn]      = (bf16)p0;
        pbuf[wv][mt * 16 + rr + 8 * hl][cn + 16] = (bf16)p1;
      }
    }
    __asm volatile("" ::: "memory");  // keep LDS store->load order (DS in-order per wave)

    // ---- reload P in A-fragment layout: lane l -> row l%16, K chunk (l/16)*16 ----
    v16bf pf[2];
#pragma unroll
    for (int mt = 0; mt < 2; ++mt) {
      v8bf plo = *(const v8bf*)&pbuf[wv][mt * 16 + r][koff];
      v8bf phi = *(const v8bf*)&pbuf[wv][mt * 16 + r][koff + 8];
#pragma unroll
      for (int i = 0; i < 8; ++i) { pf[mt][i] = plo[i]; pf[mt][i + 8] = phi[i]; }
    }

    // ---- O += P x V  (V pre-transposed -> key-contiguous B fragments) ----
    const bf16* vcol = vbase + (size_t)cn * SEQ + kb + koff;
    v16bf vf[4];
#pragma unroll
    for (int j = 0; j < 4; ++j)
      vf[j] = *(const v16bf*)(vcol + (size_t)(16 * j) * SEQ);
#pragma unroll
    for (int mt = 0; mt < 2; ++mt)
#pragma unroll
      for (int j = 0; j < 4; ++j)
        o[mt][j] = wmma_bf16(pf[mt], vf[j], o[mt][j]);
  }

  // ---- normalize and store attention output (bf16, row-major for out-proj) ----
  bf16* orow = AO + (size_t)(b * SEQ + mtile * 32) * INNER + h * HEAD_DIM;
#pragma unroll
  for (int mt = 0; mt < 2; ++mt) {
#pragma unroll
    for (int rr = 0; rr < 8; ++rr) {
      float inv = 1.0f / lrow[mt][rr];
      bf16* pr = orow + (size_t)(mt * 16 + rr + 8 * hl) * INNER;
#pragma unroll
      for (int j = 0; j < 4; ++j)
        pr[cn + 16 * j] = (bf16)(o[mt][j][rr] * inv);
    }
  }
}

// ---------------------------------------------------------------------------
// Output projection: out[8192,1024] = AO[8192,1024] x Wout[1024,1024]^T + b
// Same 64x32 wave tiling as qkv_gemm; fp32 output with bias.
// ---------------------------------------------------------------------------
__global__ void out_gemm(const bf16* __restrict__ A,    // [8192,1024]
                         const bf16* __restrict__ W,    // [1024,1024]
                         const float* __restrict__ bias,
                         float* __restrict__ C) {       // [8192,1024]
  const int K    = INNER;
  const int lane = threadIdx.x & 31;
  const int wave = threadIdx.x >> 5;
  const int m0 = blockIdx.x * 128 + (wave >> 2) * 64;
  const int n0 = blockIdx.y * 128 + (wave & 3) * 32;
  const int r    = lane & 15;
  const int hl   = lane >> 4;
  const int koff = hl * 16;

  const bf16* aB = A + (size_t)(m0 + r) * K + koff;
  const bf16* bB = W + (size_t)(n0 + r) * K + koff;

  v8f acc[4][2] = {};
  for (int k = 0; k < K; k += 32) {
    __builtin_prefetch(aB + k + 256, 0, 3);
    __builtin_prefetch(bB + k + 256, 0, 3);
    v16bf bF0 = *(const v16bf*)(bB + k);
    v16bf bF1 = *(const v16bf*)(bB + (size_t)16 * K + k);
#pragma unroll
    for (int i = 0; i < 4; ++i) {
      v16bf aF = *(const v16bf*)(aB + (size_t)(16 * i) * K + k);
      acc[i][0] = wmma_bf16(aF, bF0, acc[i][0]);
      acc[i][1] = wmma_bf16(aF, bF1, acc[i][1]);
    }
  }

  const int cn = r;
#pragma unroll
  for (int mt = 0; mt < 4; ++mt) {
#pragma unroll
    for (int nt = 0; nt < 2; ++nt) {
#pragma unroll
      for (int rr = 0; rr < 8; ++rr) {
        int m = m0 + mt * 16 + rr + 8 * hl;
        int n = n0 + nt * 16 + cn;
        C[(size_t)m * INNER + n] = acc[mt][nt][rr] + bias[n];
      }
    }
  }
}

// ---------------------------------------------------------------------------
extern "C" void kernel_launch(void* const* d_in, const int* in_sizes, int n_in,
                              void* d_out, int out_size, void* d_ws, size_t ws_size,
                              hipStream_t stream) {
  const float* x     = (const float*)d_in[0];   // [4,2048,1024]
  const float* w_qkv = (const float*)d_in[1];   // [3072,1024]
  const float* w_out = (const float*)d_in[2];   // [1024,1024]
  const float* b_out = (const float*)d_in[3];   // [1024]
  float* out = (float*)d_out;                   // [4,2048,1024]

  // workspace layout (bf16)
  bf16* p = (bf16*)d_ws;
  bf16* xb    = p; p += (size_t)TOKENS * INPUT_DIM;
  bf16* wqkvb = p; p += (size_t)3 * INNER * INPUT_DIM;
  bf16* woutb = p; p += (size_t)INNER * INNER;
  bf16* qk    = p; p += (size_t)TOKENS * 2 * INNER;   // Q|K rows
  bf16* vT    = p; p += (size_t)TOKENS * INNER;       // V transposed
  bf16* aob   = p; p += (size_t)TOKENS * INNER;

  cvt_f32_bf16<<<512, 256, 0, stream>>>(x, xb, TOKENS * INPUT_DIM);
  cvt_f32_bf16<<<512, 256, 0, stream>>>(w_qkv, wqkvb, 3 * INNER * INPUT_DIM);
  cvt_f32_bf16<<<256, 256, 0, stream>>>(w_out, woutb, INNER * INNER);

  qkv_gemm<<<dim3(TOKENS / 128, (3 * INNER) / 128), 256, 0, stream>>>(xb, wqkvb, qk, vT);

  attn_fa<<<(BATCH * HEAD_NUM * (SEQ / 32)) / 8, 256, 0, stream>>>(qk, vT, aob);

  out_gemm<<<dim3(TOKENS / 128, INNER / 128), 256, 0, stream>>>(aob, woutb, b_out, out);
}
